// NeuralPolarDecoder_7541962572026
// MI455X (gfx1250) — matrix operand order repaired and checked by reading, not compiled, not run on hardware
//
#include <hip/hip_runtime.h>
#include <hip/hip_bf16.h>

// ---------------------------------------------------------------------------
// Neural polar decoder for MI455X (gfx1250, wave32, WMMA).
// B=256, N=1024, D=64.  10 butterfly stages of two small MLPs, each lowered to
// v_wmma_f32_16x16x32_bf16 with weights resident (transposed, bf16) in LDS.
// Grid: one block per batch row (256 blocks); each block loops 4x over its
// 32 tiles so weight staging is amortized and activations stay L2-local.
// ---------------------------------------------------------------------------

#define BB   256
#define NN   1024
#define DD   64
#define NSTG 11            // stage 0 (initial) + 10 butterfly stages
#define EPSV 1e-7f

typedef __attribute__((ext_vector_type(4)))  __bf16 v4bf;
typedef __attribute__((ext_vector_type(8)))  __bf16 v8bf;
typedef __attribute__((ext_vector_type(16))) __bf16 v16bf;
typedef __attribute__((ext_vector_type(8)))  float  v8f;

// ---------------------------------------------------------------------------
// Initial embedding: e[r, :] = y[r,0]*emb_W[0,:] + y[r,1]*emb_W[1,:] + emb_b
// ---------------------------------------------------------------------------
__global__ void npd_init_e(const float* __restrict__ y,
                           const float* __restrict__ embW,
                           const float* __restrict__ embb,
                           float* __restrict__ e) {
  long idx = (long)blockIdx.x * blockDim.x + threadIdx.x;   // B*N*16 threads
  long r  = idx >> 4;
  int  dg = (int)(idx & 15) * 4;
  const float* yr = y + r * 2;
  float y0 = yr[0], y1 = yr[1];
  float4 w0 = *(const float4*)(embW + dg);
  float4 w1 = *(const float4*)(embW + DD + dg);
  float4 bv = *(const float4*)(embb + dg);
  float4 o;
  o.x = y0 * w0.x + y1 * w1.x + bv.x;
  o.y = y0 * w0.y + y1 * w1.y + bv.y;
  o.z = y0 * w0.z + y1 * w1.z + bv.z;
  o.w = y0 * w0.w + y1 * w1.w + bv.w;
  *(float4*)(e + r * DD + dg) = o;
}

// ---------------------------------------------------------------------------
// v propagation: v'[2k]=(v[i0]+v[i1])&1, v'[2k+1]=v[i1]
// ---------------------------------------------------------------------------
__global__ void npd_vprop(const int* __restrict__ v_in,
                          int* __restrict__ v_out, int lsh) {
  int p = blockIdx.x * blockDim.x + threadIdx.x;            // < B*N/2
  int b  = p >> 9;                                          // N/2 = 512
  int kk = p & 511;
  int half = 1 << lsh;
  int blk = kk >> lsh;
  int q   = kk & (half - 1);
  long base = (long)b * NN;
  long r0 = base + ((long)blk << (lsh + 1)) + q;
  long r1 = r0 + half;
  int x0 = v_in[r0], x1 = v_in[r1];
  v_out[base + 2 * kk]     = (x0 + x1) & 1;
  v_out[base + 2 * kk + 1] = x1;
}

// ---------------------------------------------------------------------------
// Softmax head + loss + norm for stage t.
// losses (B,11,N) | preds (B,N,11,2) | norms (B,11,N) concatenated in d_out.
// ---------------------------------------------------------------------------
__global__ void npd_llr(const float* __restrict__ e,
                        const int*   __restrict__ v,
                        const float* __restrict__ W,    // (D,2) row-major
                        const float* __restrict__ bias, // (2,)
                        float* __restrict__ losses,
                        float* __restrict__ preds,
                        float* __restrict__ norms, int t) {
  long r = (long)blockIdx.x * blockDim.x + threadIdx.x;    // < B*N
  const float4* er = (const float4*)(e + r * DD);
  float z0 = bias[0], z1 = bias[1], ss = 0.f;
#pragma unroll
  for (int qv = 0; qv < 16; ++qv) {
    float4 f = er[qv];
    int d = qv * 4;
    z0 += f.x * W[2 * d + 0] + f.y * W[2 * d + 2] + f.z * W[2 * d + 4] + f.w * W[2 * d + 6];
    z1 += f.x * W[2 * d + 1] + f.y * W[2 * d + 3] + f.z * W[2 * d + 5] + f.w * W[2 * d + 7];
    ss += f.x * f.x + f.y * f.y + f.z * f.z + f.w * f.w;
  }
  float mx = fmaxf(z0, z1);
  float p0 = __expf(z0 - mx), p1 = __expf(z1 - mx);
  float inv = 1.f / (p0 + p1);
  p0 *= inv; p1 *= inv;
  int vv = v[r];
  float pt = vv ? p1 : p0;
  pt = fminf(fmaxf(pt, EPSV), 1.0f);
  float loss = -__logf(pt);
  long b = r >> 10;                                        // N = 1024
  long n = r & 1023;
  long ln = (b * NSTG + t) * NN + n;
  losses[ln] = loss;
  norms[ln]  = sqrtf(ss);
  long pi = ((b * NN + n) * NSTG + t) * 2;
  preds[pi]     = p0;
  preds[pi + 1] = p1;
}

// ---------------------------------------------------------------------------
// WMMA helpers (A/B: contiguous k order chosen here, same in both; C: ISA layout):
//  A 16x32 bf16 : lanes 0-15 M=lane, K 0..15; lanes 16-31 M=lane-16, K 16..31
//                 elements j -> K=base+j
//  B 32x16 bf16 : lanes 0-15 N=lane, K 0..15; lanes 16-31 N=lane-16, K 16..31
//  C 16x16 f32  : lanes 0-15 N=lane, rows 0..7; lanes 16-31 rows 8..15
// ---------------------------------------------------------------------------
__device__ __forceinline__ v16bf npd_a_frag(const __bf16* X, int xs, int kf, int lane) {
  int rowA = lane & 15;
  int ak = (lane & 16) ? 16 : 0;
  const __bf16* p = X + rowA * xs + kf * 32 + ak;
  v8bf lo = *(const v8bf*)p;
  v8bf hi = *(const v8bf*)(p + 8);
  return __builtin_shufflevector(lo, hi, 0, 1, 2, 3, 4, 5, 6, 7,
                                 8, 9, 10, 11, 12, 13, 14, 15);
}

template <int KF>
__device__ __forceinline__ void npd_gemm(const __bf16* X, int xs,
                                         const __bf16* WT, int ws,
                                         const float* biasLds, int lane,
                                         v8f acc[4]) {
  int colid = lane & 15;
  int bk = (lane & 16) ? 16 : 0;
#pragma unroll
  for (int ct = 0; ct < 4; ++ct) {
    float bv = biasLds[ct * 16 + colid];
#pragma unroll
    for (int i = 0; i < 8; ++i) acc[ct][i] = bv;
  }
#pragma unroll
  for (int kf = 0; kf < KF; ++kf) {
    v16bf a = npd_a_frag(X, xs, kf, lane);
#pragma unroll
    for (int ct = 0; ct < 4; ++ct) {
      v16bf bmat = *(const v16bf*)(WT + (ct * 16 + colid) * ws + kf * 32 + bk);
      acc[ct] = __builtin_amdgcn_wmma_f32_16x16x32_bf16(
          false, a, false, bmat, (short)0, acc[ct], false, false);
    }
  }
}

__device__ __forceinline__ void npd_store_h(__bf16* H, int lane, const v8f acc[4]) {
  int colid = lane & 15;
  int mb = (lane & 16) ? 8 : 0;
#pragma unroll
  for (int ct = 0; ct < 4; ++ct)
#pragma unroll
    for (int i = 0; i < 8; ++i)
      H[(mb + i) * 64 + ct * 16 + colid] = (__bf16)fmaxf(acc[ct][i], 0.0f);
}

__device__ __forceinline__ void npd_store_out(float* e_out, long bRow, int kkb,
                                              int sel, int lane, const v8f acc[4]) {
  int colid = lane & 15;
  int mb = (lane & 16) ? 8 : 0;
#pragma unroll
  for (int ct = 0; ct < 4; ++ct)
#pragma unroll
    for (int i = 0; i < 8; ++i) {
      long r = bRow + 2 * (long)(kkb + mb + i) + sel;
      e_out[r * DD + ct * 16 + colid] = acc[ct][i];
    }
}

// ---------------------------------------------------------------------------
// One butterfly stage.  256 threads = 8 waves.  Block b owns batch row b
// (32 tiles); 4 iterations x 8 waves, one 16-pair tile per wave per iter.
// Dynamic LDS layout (bytes):
//   WT1L 64x128 bf16 (16384) | WT2L 64x64 (8192) | WT1R 64x192 (24576)
//   WT2R 64x64 (8192) | biases 4x64 f32 (1024) | X 8x16x192 bf16 (49152)
//   H 8x16x64 bf16 (16384)                                   total 123904
// ---------------------------------------------------------------------------
__global__ void __launch_bounds__(256)
npd_stage(const float* __restrict__ e_in, const int* __restrict__ v_in,
          float* __restrict__ e_out, const float* __restrict__ lab_emb,
          const float* __restrict__ cnW1, const float* __restrict__ cnW2,
          const float* __restrict__ bnW1, const float* __restrict__ bnW2,
          const float* __restrict__ cnb1, const float* __restrict__ cnb2,
          const float* __restrict__ bnb1, const float* __restrict__ bnb2,
          int lsh) {
  extern __shared__ char smem[];
  __bf16* WT1L = (__bf16*)smem;                  // [c*128 + k]
  __bf16* WT2L = WT1L + 64 * 128;                // [c*64 + k]
  __bf16* WT1R = WT2L + 64 * 64;                 // [c*192 + k]
  __bf16* WT2R = WT1R + 64 * 192;                // [c*64 + k]
  float*  biasL = (float*)(WT2R + 64 * 64);      // [b1L|b2L|b1R|b2R]
  __bf16* Xall = (__bf16*)(biasL + 256);
  __bf16* Hall = Xall + 8 * 16 * 192;

  int tid = threadIdx.x;
  int lane = tid & 31;
  int wid = tid >> 5;

  // --- cooperative weight staging: f32 -> bf16, transposed (once/block) --
  for (int i = tid; i < 64 * 128; i += 256)
    WT1L[i] = (__bf16)cnW1[(i & 127) * 64 + (i >> 7)];
  for (int i = tid; i < 64 * 64; i += 256)
    WT2L[i] = (__bf16)cnW2[(i & 63) * 64 + (i >> 6)];
  for (int i = tid; i < 64 * 192; i += 256)
    WT1R[i] = (__bf16)bnW1[(i % 192) * 64 + (i / 192)];
  for (int i = tid; i < 64 * 64; i += 256)
    WT2R[i] = (__bf16)bnW2[(i & 63) * 64 + (i >> 6)];
  if (tid < 64)        biasL[tid] = cnb1[tid];
  else if (tid < 128)  biasL[tid] = cnb2[tid - 64];
  else if (tid < 192)  biasL[tid] = bnb1[tid - 128];
  else                 biasL[tid] = bnb2[tid - 192];
  __syncthreads();

  int  b    = blockIdx.x;                        // one batch row per block
  long bRow = (long)b * NN;
  __bf16* Xw = Xall + wid * (16 * 192);
  __bf16* Hw = Hall + wid * (16 * 64);

  int m  = lane >> 1;                            // staging row 0..15
  int ch = (lane & 1) * 32;                      // staging column half
  int half = 1 << lsh;

  for (int it = 0; it < 4; ++it) {
    int kkb = (it * 8 + wid) << 4;               // pair base within (N/2)

    // --- stage X = [e_odd | e_even | lab_emb[xor]] as 16x192 bf16 --------
    {
      int kk = kkb + m;
      int blk = kk >> lsh;
      int q   = kk & (half - 1);
      long r0 = bRow + ((long)blk << (lsh + 1)) + q;
      long r1 = r0 + half;
      int vx = (v_in[r0] + v_in[r1]) & 1;
      const float* s0 = e_in + r0 * DD + ch;
      const float* s1 = e_in + r1 * DD + ch;
      const float* s2 = lab_emb + (long)vx * DD + ch;
      __bf16* xr = Xw + m * 192;
#pragma unroll
      for (int j = 0; j < 32; j += 4) {
        float4 f0 = *(const float4*)(s0 + j);
        float4 f1 = *(const float4*)(s1 + j);
        float4 f2 = *(const float4*)(s2 + j);
        v4bf o0, o1, o2;
        o0[0] = (__bf16)f0.x; o0[1] = (__bf16)f0.y; o0[2] = (__bf16)f0.z; o0[3] = (__bf16)f0.w;
        o1[0] = (__bf16)f1.x; o1[1] = (__bf16)f1.y; o1[2] = (__bf16)f1.z; o1[3] = (__bf16)f1.w;
        o2[0] = (__bf16)f2.x; o2[1] = (__bf16)f2.y; o2[2] = (__bf16)f2.z; o2[3] = (__bf16)f2.w;
        *(v4bf*)(xr + ch + j)       = o0;
        *(v4bf*)(xr + 64 + ch + j)  = o1;
        *(v4bf*)(xr + 128 + ch + j) = o2;
      }
    }
    __syncthreads();

    v8f acc[4];

    // --- left MLP: relu(X[:, :128] @ W1L + b1L) @ W2L + b2L -> rows 2k ---
    npd_gemm<4>(Xw, 192, WT1L, 128, biasL, lane, acc);
    npd_store_h(Hw, lane, acc);
    __syncthreads();
    npd_gemm<2>(Hw, 64, WT2L, 64, biasL + 64, lane, acc);
    npd_store_out(e_out, bRow, kkb, /*sel=*/0, lane, acc);
    __syncthreads();

    // --- right MLP: relu(X @ W1R + b1R) @ W2R + b2R -> rows 2k+1 ---------
    npd_gemm<6>(Xw, 192, WT1R, 192, biasL + 128, lane, acc);
    npd_store_h(Hw, lane, acc);
    __syncthreads();
    npd_gemm<2>(Hw, 64, WT2R, 64, biasL + 192, lane, acc);
    npd_store_out(e_out, bRow, kkb, /*sel=*/1, lane, acc);
    __syncthreads();
  }
}

// ---------------------------------------------------------------------------
// Host-side launcher
// ---------------------------------------------------------------------------
extern "C" void kernel_launch(void* const* d_in, const int* in_sizes, int n_in,
                              void* d_out, int out_size, void* d_ws, size_t ws_size,
                              hipStream_t stream) {
  (void)in_sizes; (void)n_in; (void)out_size; (void)ws_size;
  const int*   x     = (const int*)  d_in[0];
  const float* y     = (const float*)d_in[1];
  const float* embW  = (const float*)d_in[2];
  const float* embb  = (const float*)d_in[3];
  const float* lab   = (const float*)d_in[4];
  const float* cnW1  = (const float*)d_in[5];
  const float* cnb1  = (const float*)d_in[6];
  const float* cnW2  = (const float*)d_in[7];
  const float* cnb2  = (const float*)d_in[8];
  const float* bnW1  = (const float*)d_in[9];
  const float* bnb1  = (const float*)d_in[10];
  const float* bnW2  = (const float*)d_in[11];
  const float* bnb2  = (const float*)d_in[12];
  const float* llrW  = (const float*)d_in[13];
  const float* llrb  = (const float*)d_in[14];

  float* losses = (float*)d_out;                         // B*11*N
  float* preds  = losses + (size_t)BB * NSTG * NN;       // B*N*11*2
  float* norms  = preds + (size_t)BB * NN * NSTG * 2;    // B*11*N

  // workspace: e ping-pong (64 MB each) + v ping-pong (1 MB each)
  float* e0 = (float*)d_ws;
  float* e1 = e0 + (size_t)BB * NN * DD;
  int*   v0 = (int*)(e1 + (size_t)BB * NN * DD);
  int*   v1 = v0 + (size_t)BB * NN;
  float* ebuf[2] = {e0, e1};
  int*   vbuf[2] = {v0, v1};

  const size_t smem = 123904;

  npd_init_e<<<(BB * NN * 16) / 256, 256, 0, stream>>>(y, embW, embb, e0);
  npd_llr<<<(BB * NN) / 256, 256, 0, stream>>>(e0, x, llrW, llrb,
                                               losses, preds, norms, 0);

  const float* ein = e0;
  const int*   vin = x;
  for (int t = 1; t <= 10; ++t) {
    float* eout = ebuf[t & 1];
    int*   vout = vbuf[t & 1];
    int lsh = t - 1;
    npd_stage<<<BB, 256, smem, stream>>>(ein, vin, eout, lab,
                                         cnW1, cnW2, bnW1, bnW2,
                                         cnb1, cnb2, bnb1, bnb2, lsh);
    npd_vprop<<<(BB * NN / 2) / 256, 256, 0, stream>>>(vin, vout, lsh);
    npd_llr<<<(BB * NN) / 256, 256, 0, stream>>>(eout, vout, llrW, llrb,
                                                 losses, preds, norms, t);
    ein = eout;
    vin = vout;
  }
}
